// ScaledDotProductAttention_91122026152269
// MI455X (gfx1250) — matrix operand order repaired and checked by reading, not compiled, not run on hardware
//
#include <hip/hip_runtime.h>
#include <hip/hip_bf16.h>

#define BATCH 8
#define SEQ   2048
#define MDIM  1024
#define DKV   128

typedef __attribute__((ext_vector_type(8)))  float          v8f;
typedef __attribute__((ext_vector_type(8)))  unsigned short v8u;
typedef __attribute__((ext_vector_type(16))) unsigned short v16u;
typedef __attribute__((ext_vector_type(16))) __bf16         v16bf;
typedef __attribute__((ext_vector_type(4)))  unsigned int   u32x4;
typedef __attribute__((ext_vector_type(8)))  int            i32x8;
typedef __attribute__((ext_vector_type(4)))  int            i32x4;

#define AS1 __attribute__((address_space(1)))
#define AS3 __attribute__((address_space(3)))

#if __has_builtin(__builtin_amdgcn_global_load_async_to_lds_b128) && \
    __has_builtin(__builtin_amdgcn_s_wait_asynccnt)
#define HAVE_ASYNC_LDS 1
#else
#define HAVE_ASYNC_LDS 0
#endif

#if __has_builtin(__builtin_amdgcn_tensor_load_to_lds) && \
    __has_builtin(__builtin_amdgcn_s_wait_tensorcnt)
#define HAVE_TDM 1
#else
#define HAVE_TDM 0
#endif

static __device__ __forceinline__ unsigned short f2bf(float f) {
  union { float f; unsigned int u; } v; v.f = f;
  unsigned int r = v.u + 0x7FFFu + ((v.u >> 16) & 1u);   // round-to-nearest-even
  return (unsigned short)(r >> 16);
}

static __device__ __forceinline__ v8f wmma_bf16(v16u a, v16u b, v8f c) {
  return __builtin_amdgcn_wmma_f32_16x16x32_bf16(
      false, __builtin_bit_cast(v16bf, a),
      false, __builtin_bit_cast(v16bf, b),
      (short)0, c, false, false);
}

static __device__ __forceinline__ v16u pack16(v8u lo, v8u hi) {
  v16u r;
#pragma unroll
  for (int i = 0; i < 8; ++i) { r[i] = lo[i]; r[i + 8] = hi[i]; }
  return r;
}

// ---------------------------------------------------------------------------
// Kernel 1: weights f32 [M x 128] -> bf16 transposed [128 x M]
// ---------------------------------------------------------------------------
__global__ void wconv_kernel(const float* __restrict__ Wq,
                             const float* __restrict__ Wk,
                             const float* __restrict__ Wv,
                             unsigned short* __restrict__ wT) {
  int idx = blockIdx.x * 256 + threadIdx.x;
  if (idx >= 3 * DKV * MDIM) return;
  int w   = idx / (DKV * MDIM);
  int rem = idx - w * (DKV * MDIM);
  int n = rem >> 10;
  int m = rem & (MDIM - 1);
  const float* src = (w == 0) ? Wq : (w == 1) ? Wk : Wv;
  wT[idx] = f2bf(src[m * DKV + n]);
}

// ---------------------------------------------------------------------------
// Kernel 2: projections; input strip staged via async-to-LDS, then converted.
// ---------------------------------------------------------------------------
__global__ __launch_bounds__(256)
void proj_kernel(const float* __restrict__ Xq, const float* __restrict__ Xk,
                 const float* __restrict__ Xv,
                 const unsigned short* __restrict__ wT,
                 unsigned short* __restrict__ qb, unsigned short* __restrict__ kb,
                 unsigned short* __restrict__ vbT) {
  __shared__ __align__(16) float          Asf[16 * MDIM];   // raw f32 strip (64 KB)
  __shared__ __align__(16) unsigned short As[16 * 1032];    // bf16, padded rows

  const int which = blockIdx.y;
  const int tile  = blockIdx.x;
  const int b  = tile >> 7;
  const int s0 = (tile & 127) << 4;
  const float* src = ((which == 0) ? Xq : (which == 1) ? Xk : Xv)
                     + (size_t)(b * SEQ + s0) * MDIM;

  // stage 16x1024 f32 into LDS
#if HAVE_ASYNC_LDS
  for (int i = threadIdx.x; i < 16 * MDIM / 4; i += 256) {
    __builtin_amdgcn_global_load_async_to_lds_b128(
        (AS1 i32x4*)(src + i * 4), (AS3 i32x4*)&Asf[i * 4], 0, 0);
  }
  __builtin_amdgcn_s_wait_asynccnt(0);
#else
  for (int i = threadIdx.x; i < 16 * MDIM / 4; i += 256)
    ((float4*)Asf)[i] = ((const float4*)src)[i];
#endif
  __syncthreads();

  // convert to bf16 with padded row stride (1032) for conflict-free frag reads
  for (int i = threadIdx.x; i < 16 * MDIM / 4; i += 256) {
    float4 f = ((const float4*)Asf)[i];
    int row = i >> 8;
    int c   = (i & 255) << 2;
    unsigned short* d = &As[row * 1032 + c];
    d[0] = f2bf(f.x); d[1] = f2bf(f.y); d[2] = f2bf(f.z); d[3] = f2bf(f.w);
  }
  __syncthreads();

  const int wv = threadIdx.x >> 5;
  const int l  = threadIdx.x & 31;
  const int hl = l >> 4;
  const int lm = l & 15;

  v8f acc = {};
  const unsigned short* wrow = wT + which * (DKV * MDIM) + (wv * 16 + lm) * MDIM;
#pragma unroll 4
  for (int kk = 0; kk < MDIM; kk += 32) {
    const unsigned short* ar = &As[lm * 1032 + kk + hl * 16];
    v16u a  = pack16(*(const v8u*)ar, *(const v8u*)(ar + 8));
    v16u bf = *(const v16u*)(wrow + kk + hl * 16);
    acc = wmma_bf16(a, bf, acc);
  }

  if (which < 2) {
    unsigned short* dst = (which == 0 ? qb : kb)
        + (size_t)(b * SEQ + s0) * DKV + wv * 16 + lm;
#pragma unroll
    for (int r = 0; r < 8; ++r)
      dst[(size_t)(r + 8 * hl) * DKV] = f2bf(acc[r]);
  } else {
    unsigned short tmp[8];
#pragma unroll
    for (int r = 0; r < 8; ++r) tmp[r] = f2bf(acc[r]);
    unsigned short* dst = vbT + (size_t)(b * DKV + wv * 16 + lm) * SEQ + s0 + 8 * hl;
    *(uint4*)dst = *(const uint4*)tmp;
  }
}

// ---------------------------------------------------------------------------
// Kernel 3: causal flash attention; K chunks fetched by the Tensor Data Mover
// into double-buffered LDS (padded rows: 272B stride), waited on TENSORcnt.
// ---------------------------------------------------------------------------
#define KROW 136   // ushorts per padded K row in LDS (128 data + 8 pad)

#if HAVE_TDM
static __device__ __forceinline__ void tdm_load_k(const unsigned short* gsrc,
                                                  unsigned lds_off) {
  unsigned long long ga = (unsigned long long)gsrc;
  u32x4 g0;
  g0[0] = 1u;                                            // count=1, user mode
  g0[1] = lds_off;                                       // LDS byte address
  g0[2] = (unsigned)ga;                                  // global addr lo
  g0[3] = (unsigned)((ga >> 32) & 0x1FFFFFFu) | (2u << 30); // addr hi | type=2
  i32x8 g1;
  // data_size=2B | pad_enable | pad_interval=64dw | pad_amount=4dw
  g1[0] = (1 << 16) | (1 << 20) | (5 << 22) | (3 << 25);
  g1[1] = (int)(128u << 16);   // tensor_dim0 = 128 (low16 in bits[63:48])
  g1[2] = (int)(32u << 16);    // tensor_dim1 = 32
  g1[3] = (int)(128u << 16);   // tile_dim0 = 128
  g1[4] = 32;                  // tile_dim1 = 32, tile_dim2 = 0
  g1[5] = 128;                 // tensor_dim0_stride = 128
  g1[6] = 0;
  g1[7] = 0;
  i32x4 g2 = {0, 0, 0, 0};
  i32x4 g3 = {0, 0, 0, 0};
#if __has_include(<hip/amd_detail/amd_gfx1250_TDM.h>)
  i32x8 g4 = {0, 0, 0, 0, 0, 0, 0, 0};
  __builtin_amdgcn_tensor_load_to_lds(g0, g1, g2, g3, g4, 0);
#else
  __builtin_amdgcn_tensor_load_to_lds(g0, g1, g2, g3, 0);
#endif
}
#endif

__global__ __launch_bounds__(128)
void attn_kernel(const unsigned short* __restrict__ qb,
                 const unsigned short* __restrict__ kb,
                 const unsigned short* __restrict__ vbT,
                 float* __restrict__ out) {
#if HAVE_TDM
  __shared__ __align__(16) unsigned short Kbuf[4][2][32 * KROW]; // 68 KB
#endif
  __shared__ __align__(16) unsigned short Pl[4][16 * 40];
  const int wv = threadIdx.x >> 5;
  const int l  = threadIdx.x & 31;
  const int hl = l >> 4;
  const int lm = l & 15;
  const int tile = blockIdx.x * 4 + wv;
  const int b  = tile >> 7;
  const int q0 = (tile & 127) << 4;
  unsigned short* P = &Pl[wv][0];

  // Q fragments, resident for the whole scan
  const unsigned short* qrow = qb + (size_t)(b * SEQ + q0 + lm) * DKV;
  v16u qa[4];
#pragma unroll
  for (int c = 0; c < 4; ++c) {
    const unsigned short* p = qrow + c * 32 + hl * 16;
    qa[c] = pack16(*(const v8u*)p, *(const v8u*)(p + 8));
  }

  v8f zero = {};
  v8f o[8];
#pragma unroll
  for (int t = 0; t < 8; ++t) o[t] = zero;
  float rmax[8], rsum[8];
#pragma unroll
  for (int r = 0; r < 8; ++r) { rmax[r] = -3.0e38f; rsum[r] = 0.0f; }

  const float sc = 0.08838834764831845f * 1.4426950408889634f;  // 1/sqrt(dk)*log2e
  const int kend = q0 + 16;

#if HAVE_TDM
  unsigned koff[2];
#pragma unroll
  for (int i = 0; i < 2; ++i)
    koff[i] = (unsigned)(unsigned long long)(AS3 char*)&Kbuf[wv][i][0];
  tdm_load_k(kb + (size_t)(b * SEQ) * DKV, koff[0]);   // prologue: chunk 0
  int cur = 0;
#endif

  for (int j0 = 0; j0 < kend; j0 += 32) {
    __builtin_prefetch(vbT + (size_t)(b * DKV + lm) * SEQ + j0, 0, 3);

#if HAVE_TDM
    if (j0 + 32 < kend) {
      tdm_load_k(kb + (size_t)(b * SEQ + j0 + 32) * DKV, koff[cur ^ 1]);
      __builtin_amdgcn_s_wait_tensorcnt(1);   // in-order: current chunk resident
    } else {
      __builtin_amdgcn_s_wait_tensorcnt(0);
    }
    const unsigned short* kr0 = &Kbuf[wv][cur][lm * KROW + hl * 16];
    const unsigned short* kr1 = &Kbuf[wv][cur][(16 + lm) * KROW + hl * 16];
    cur ^= 1;
#else
    const unsigned short* kr0 = kb + (size_t)(b * SEQ + j0 + lm) * DKV + hl * 16;
    const unsigned short* kr1 = kr0 + 16 * DKV;
#endif

    v8f s0 = zero, s1 = zero;
#pragma unroll
    for (int c = 0; c < 4; ++c) s0 = wmma_bf16(qa[c], *(const v16u*)(kr0 + c * 32), s0);
#pragma unroll
    for (int c = 0; c < 4; ++c) s1 = wmma_bf16(qa[c], *(const v16u*)(kr1 + c * 32), s1);

    const bool needmask = (j0 + 31) > q0;
    float corr[8];
#pragma unroll
    for (int r = 0; r < 8; ++r) {
      float v0 = s0[r] * sc;
      float v1 = s1[r] * sc;
      if (needmask) {
        int m = q0 + r + 8 * hl;
        if (j0 + lm      > m) v0 = -3.0e38f;
        if (j0 + 16 + lm > m) v1 = -3.0e38f;
      }
      float tm = fmaxf(v0, v1);
#pragma unroll
      for (int d = 1; d < 16; d <<= 1) tm = fmaxf(tm, __shfl_xor(tm, d, 32));
      float nm = fmaxf(rmax[r], tm);
      corr[r]  = exp2f(rmax[r] - nm);
      rmax[r]  = nm;
      float p0 = exp2f(v0 - nm);
      float p1 = exp2f(v1 - nm);
      float ps = p0 + p1;
#pragma unroll
      for (int d = 1; d < 16; d <<= 1) ps += __shfl_xor(ps, d, 32);
      rsum[r] = rsum[r] * corr[r] + ps;
      P[(r + 8 * hl) * 40 + lm]      = f2bf(p0);
      P[(r + 8 * hl) * 40 + 16 + lm] = f2bf(p1);
    }
#pragma unroll
    for (int t = 0; t < 8; ++t)
#pragma unroll
      for (int r = 0; r < 8; ++r) o[t][r] *= corr[r];

    const unsigned short* pr = &P[lm * 40 + hl * 16];
    v16u pa = pack16(*(const v8u*)pr, *(const v8u*)(pr + 8));

#pragma unroll
    for (int t = 0; t < 8; ++t) {
      const unsigned short* vr =
          vbT + (size_t)(b * DKV + t * 16 + lm) * SEQ + j0 + hl * 16;
      o[t] = wmma_bf16(pa, *(const v16u*)vr, o[t]);
    }
  }

  float* dst = out + (size_t)(b * SEQ + q0) * DKV;
#pragma unroll
  for (int r = 0; r < 8; ++r) {
    float inv = 1.0f / rsum[r];
#pragma unroll
    for (int t = 0; t < 8; ++t)
      dst[(size_t)(r + 8 * hl) * DKV + t * 16 + lm] = o[t][r] * inv;
  }
}

// ---------------------------------------------------------------------------
extern "C" void kernel_launch(void* const* d_in, const int* in_sizes, int n_in,
                              void* d_out, int out_size, void* d_ws, size_t ws_size,
                              hipStream_t stream) {
  const float* query = (const float*)d_in[0];
  const float* key   = (const float*)d_in[1];
  const float* value = (const float*)d_in[2];
  const float* Wq    = (const float*)d_in[3];
  const float* Wk    = (const float*)d_in[4];
  const float* Wv    = (const float*)d_in[5];
  float* out = (float*)d_out;

  unsigned short* wT  = (unsigned short*)d_ws;
  unsigned short* qb  = wT + 3 * DKV * MDIM;
  unsigned short* kb  = qb + (size_t)BATCH * SEQ * DKV;
  unsigned short* vbT = kb + (size_t)BATCH * SEQ * DKV;

  wconv_kernel<<<(3 * DKV * MDIM + 255) / 256, 256, 0, stream>>>(Wq, Wk, Wv, wT);
  proj_kernel<<<dim3(BATCH * SEQ / 16, 3), 256, 0, stream>>>(query, key, value,
                                                             wT, qb, kb, vbT);
  attn_kernel<<<BATCH * SEQ / 16 / 4, 128, 0, stream>>>(qb, kb, vbT, out);

  (void)in_sizes; (void)n_in; (void)out_size; (void)ws_size;
}
